// HierarchicalMemoryMoE_3032246911412
// MI455X (gfx1250) — matrix operand order repaired and checked by reading, not compiled, not run on hardware
//
#include <hip/hip_runtime.h>

#define E_   4
#define N_   2048
#define D_   512
#define H_   8
#define DH_  64
#define DFF_ 2048
#define M_   2048
#define L_   3
#define CTX_ 64

#define USE_ASYNC_LDS 1

typedef unsigned short u16;
typedef __attribute__((ext_vector_type(16))) __bf16 v16bf;
typedef __attribute__((ext_vector_type(8)))  float  v8f;
typedef unsigned int v4u __attribute__((ext_vector_type(4)));

union Frag { v16bf v; v4u q[2]; };

__device__ __forceinline__ u16 f2bf(float f) {
  unsigned u = __builtin_bit_cast(unsigned, f);
  unsigned r = 0x7FFFu + ((u >> 16) & 1u);   // round-to-nearest-even
  return (u16)((u + r) >> 16);
}

enum { F_CLIP = 1, F_RELU = 2 };

#define BM 256
#define BN 128
#define BK 32
#define LDT 40   // padded LDS row stride (u16): 80B -> conflict-free 16B fragment reads

#if USE_ASYNC_LDS
// One 16B async DMA: global -> LDS, per lane. Tracked by ASYNCcnt.
__device__ __forceinline__ void async_ld16(const u16* g, unsigned lds_off) {
  asm volatile("global_load_async_to_lds_b128 %0, %1, off"
               :: "v"(lds_off), "v"(g) : "memory");
}
__device__ __forceinline__ void wait_async0() {
  asm volatile("s_wait_asynccnt 0x0" ::: "memory");
}
#endif

// ---------------------------------------------------------------------------
// bf16 WMMA GEMM:  C[M,N] = post( alpha * A[M,K] @ B[N,K]^T + bias + resid )
// A,B row-major bf16 (u16), C fp32. 256 threads = 8 wave32s.
// Block tile 256x128, wave tile 64x64 (4x4 WMMA 16x16x32), K-step 32,
// double-buffered LDS fed by async global->LDS DMA.
// REQUIRES: M % 256 == 0; K % 32 == 0; B readable for ceil(N/128)*128 rows.
// ---------------------------------------------------------------------------
__global__ __launch_bounds__(256)
void wmma_gemm(const u16* __restrict__ A, int lda,
               const u16* __restrict__ B, int ldb,
               float* __restrict__ C, int ldc,
               const float* __restrict__ bias,
               const float* __restrict__ resid, int ldr,
               int M, int N, int K, float alpha, int flags)
{
  __shared__ __align__(16) u16 sA[2][BM * LDT];
  __shared__ __align__(16) u16 sB[2][BN * LDT];

  const int tid  = threadIdx.x;
  const int lane = tid & 31;
  const int wave = tid >> 5;
  const int wr   = wave >> 1;        // 0..3 : 64-row group
  const int wc   = wave & 1;         // 0..1 : 64-col group
  const int lr   = lane & 15;
  const int hi   = (lane >> 4) & 1;

  const int bm = blockIdx.y * BM;
  const int bn = blockIdx.x * BN;

  // staging map: A -> thread t owns full row t (64B); B -> thread t owns half row
  const int brow = tid >> 1;
  const int bcol = (tid & 1) << 4;

  const u16* gA = A + (size_t)(bm + tid) * lda;          // + k0 per step
  const u16* gB = B + (size_t)(bn + brow) * ldb + bcol;  // + k0 per step

  const v8f zero8 = {0.f,0.f,0.f,0.f,0.f,0.f,0.f,0.f};
  v8f acc[4][4];
  #pragma unroll
  for (int i = 0; i < 4; ++i)
    #pragma unroll
    for (int j = 0; j < 4; ++j)
      acc[i][j] = zero8;

  // K-chunk offsets used here (contiguous order, shared by A and B):
  // A (16x32): lane<16 -> K {0..15};        lane>=16 -> K {16..31}
  // B (32x16): lane<16 -> K {0..15};        lane>=16 -> K {16..31}
  const int ka0 = hi ? 16 : 0;
  const int ka1 = hi ? 24 : 8;
  const int kb0 = hi ? 16 : 0;

  const int nk = K >> 5;

#if USE_ASYNC_LDS
  {
    unsigned la = (unsigned)(size_t)&sA[0][tid * LDT];
    #pragma unroll
    for (int c = 0; c < 4; ++c) async_ld16(gA + c * 8, la + c * 16);
    unsigned lb = (unsigned)(size_t)&sB[0][brow * LDT + bcol];
    async_ld16(gB,     lb);
    async_ld16(gB + 8, lb + 16);
  }
  for (int k = 0; k < nk; ++k) {
    wait_async0();          // this wave's DMA for tile k complete
    __syncthreads();        // everyone's DMA complete; prev-buffer reads done
    if (k + 1 < nk) {
      const int kn = (k + 1) << 5;
      const int nb = (k + 1) & 1;
      unsigned la = (unsigned)(size_t)&sA[nb][tid * LDT];
      #pragma unroll
      for (int c = 0; c < 4; ++c) async_ld16(gA + kn + c * 8, la + c * 16);
      unsigned lb = (unsigned)(size_t)&sB[nb][brow * LDT + bcol];
      async_ld16(gB + kn,     lb);
      async_ld16(gB + kn + 8, lb + 16);
    }
#else
  for (int k = 0; k < nk; ++k) {
    {
      const int k0 = k << 5;
      const u16* pa = gA + k0;
      v4u a0 = *(const v4u*)(pa), a1 = *(const v4u*)(pa + 8);
      v4u a2 = *(const v4u*)(pa + 16), a3 = *(const v4u*)(pa + 24);
      const u16* pb = gB + k0;
      v4u b0 = *(const v4u*)(pb), b1 = *(const v4u*)(pb + 8);
      u16* da = &sA[k & 1][tid * LDT];
      *(v4u*)(da) = a0; *(v4u*)(da + 8) = a1; *(v4u*)(da + 16) = a2; *(v4u*)(da + 24) = a3;
      u16* db = &sB[k & 1][brow * LDT + bcol];
      *(v4u*)(db) = b0; *(v4u*)(db + 8) = b1;
    }
    __syncthreads();
#endif
    const u16* sAb = sA[k & 1];
    const u16* sBb = sB[k & 1];

    Frag fb[4];
    #pragma unroll
    for (int ct = 0; ct < 4; ++ct) {
      const u16* p = &sBb[(wc * 64 + ct * 16 + lr) * LDT + kb0];
      fb[ct].q[0] = *(const v4u*)(p);
      fb[ct].q[1] = *(const v4u*)(p + 8);
    }
    #pragma unroll
    for (int rt = 0; rt < 4; ++rt) {
      Frag fa;
      const u16* p = &sAb[(wr * 64 + rt * 16 + lr) * LDT];
      fa.q[0] = *(const v4u*)(p + ka0);
      fa.q[1] = *(const v4u*)(p + ka1);
      #pragma unroll
      for (int ct = 0; ct < 4; ++ct) {
        acc[rt][ct] = __builtin_amdgcn_wmma_f32_16x16x32_bf16(
            false, fa.v, false, fb[ct].v, (short)0, acc[rt][ct], false, false);
      }
    }
#if !USE_ASYNC_LDS
    __syncthreads();
#endif
  }

  // C/D layout: VGPR r -> M = r + 8*hi, N = lane&15
  #pragma unroll
  for (int rt = 0; rt < 4; ++rt) {
    #pragma unroll
    for (int ct = 0; ct < 4; ++ct) {
      int n = bn + wc * 64 + ct * 16 + lr;
      if (n >= N) continue;
      #pragma unroll
      for (int r = 0; r < 8; ++r) {
        int m = bm + wr * 64 + rt * 16 + r + (hi ? 8 : 0);
        float v = acc[rt][ct][r] * alpha;
        if (flags & F_CLIP) v = fminf(fmaxf(v, -5.0f), 5.0f);
        if (bias)  v += bias[n];
        if (resid) v += resid[(size_t)m * ldr + n];
        if (flags & F_RELU) v = fmaxf(v, 0.0f);
        C[(size_t)m * ldc + n] = v;
      }
    }
  }
}

// ---------------------------------------------------------------------------
// Row LayerNorm, D=512 fixed, one block (256 thr) per row. Optional affine.
// ---------------------------------------------------------------------------
__global__ __launch_bounds__(256)
void ln_rows(float* __restrict__ outf, u16* __restrict__ outb,
             const float* __restrict__ x,
             const float* __restrict__ gamma, const float* __restrict__ beta)
{
  __shared__ float red[256];
  const int row = blockIdx.x, t = threadIdx.x;
  const float* xr = x + (size_t)row * D_;
  float v0 = xr[t], v1 = xr[t + 256];
  red[t] = v0 + v1; __syncthreads();
  for (int s = 128; s > 0; s >>= 1) { if (t < s) red[t] += red[t + s]; __syncthreads(); }
  float mean = red[0] * (1.0f / D_); __syncthreads();
  float d0 = v0 - mean, d1 = v1 - mean;
  red[t] = d0 * d0 + d1 * d1; __syncthreads();
  for (int s = 128; s > 0; s >>= 1) { if (t < s) red[t] += red[t + s]; __syncthreads(); }
  float rs = rsqrtf(red[0] * (1.0f / D_) + 1e-5f);
  float y0 = d0 * rs, y1 = d1 * rs;
  if (gamma) { y0 = y0 * gamma[t] + beta[t]; y1 = y1 * gamma[t + 256] + beta[t + 256]; }
  if (outf) { outf[(size_t)row * D_ + t] = y0; outf[(size_t)row * D_ + t + 256] = y1; }
  if (outb) { outb[(size_t)row * D_ + t] = f2bf(y0); outb[(size_t)row * D_ + t + 256] = f2bf(y1); }
}

// Row softmax: fp32 in -> bf16 probs out. cols <= 2048. Block (256) per row.
__global__ __launch_bounds__(256)
void softmax_rows(u16* __restrict__ P, const float* __restrict__ S, int cols)
{
  __shared__ float red[256];
  const int row = blockIdx.x, t = threadIdx.x;
  const float* sr = S + (size_t)row * cols;
  float v[8];
  const int cnt = (cols + 255) >> 8;
  float mx = -1e30f;
  for (int i = 0; i < cnt; ++i) {
    int c = t + (i << 8);
    v[i] = (c < cols) ? sr[c] : -1e30f;
    mx = fmaxf(mx, v[i]);
  }
  red[t] = mx; __syncthreads();
  for (int s = 128; s > 0; s >>= 1) { if (t < s) red[t] = fmaxf(red[t], red[t + s]); __syncthreads(); }
  mx = red[0]; __syncthreads();
  float sum = 0.f;
  for (int i = 0; i < cnt; ++i) {
    int c = t + (i << 8);
    float e = (c < cols) ? __expf(v[i] - mx) : 0.f;
    v[i] = e; sum += e;
  }
  red[t] = sum; __syncthreads();
  for (int s = 128; s > 0; s >>= 1) { if (t < s) red[t] += red[t + s]; __syncthreads(); }
  float inv = 1.0f / red[0];
  u16* pr = P + (size_t)row * cols;
  for (int i = 0; i < cnt; ++i) {
    int c = t + (i << 8);
    if (c < cols) pr[c] = f2bf(v[i] * inv);
  }
}

__global__ void cvt_bf16(u16* __restrict__ dst, const float* __restrict__ src, int n) {
  int i = blockIdx.x * 256 + threadIdx.x;
  if (i < n) dst[i] = f2bf(src[i]);
}

__global__ void pack_cols(u16* __restrict__ dst, const float* __restrict__ src,
                          int rows, int cols, int sstride) {
  int i = blockIdx.x * 256 + threadIdx.x;
  if (i < rows * cols) {
    int r = i / cols, c = i - r * cols;
    dst[i] = f2bf(src[(size_t)r * sstride + c]);
  }
}

__global__ void transpose_pack(u16* __restrict__ dst, const float* __restrict__ src,
                               int rows, int cols, int sstride) {
  int i = blockIdx.x * 256 + threadIdx.x;
  if (i < rows * cols) {
    int r = i / cols, c = i - r * cols;
    dst[(size_t)c * rows + r] = f2bf(src[(size_t)r * sstride + c]);
  }
}

__global__ void mean_pool(float* __restrict__ pooled, const float* __restrict__ outs) {
  int idx = blockIdx.x * 256 + threadIdx.x;            // E_*D_ threads
  int e = idx >> 9, d = idx & (D_ - 1);
  const float* p = outs + (size_t)e * N_ * D_ + d;
  float s = 0.f;
  for (int n = 0; n < N_; ++n) s += p[(size_t)n * D_];
  pooled[idx] = s * (1.0f / N_);
}

__global__ __launch_bounds__(256)
void gate_kernel(float* __restrict__ w_ws, float* __restrict__ w_out,
                 const float* __restrict__ pooled, const float* __restrict__ gate) {
  __shared__ float red[256];
  __shared__ float gl[E_];
  const int t = threadIdx.x;
  float g0 = gate[t], g1 = gate[t + 256];
  for (int e = 0; e < E_; ++e) {
    float a = pooled[e * D_ + t] * g0 + pooled[e * D_ + 256 + t] * g1;
    red[t] = a; __syncthreads();
    for (int s = 128; s > 0; s >>= 1) { if (t < s) red[t] += red[t + s]; __syncthreads(); }
    if (t == 0) gl[e] = fminf(fmaxf(red[0], -5.f), 5.f);
    __syncthreads();
  }
  if (t == 0) {
    float mx = gl[0];
    for (int e = 1; e < E_; ++e) mx = fmaxf(mx, gl[e]);
    float s = 0.f, w[E_];
    for (int e = 0; e < E_; ++e) { w[e] = __expf(gl[e] - mx); s += w[e]; }
    for (int e = 0; e < E_; ++e) { float ww = w[e] / s; w_ws[e] = ww; w_out[e] = ww; }
  }
}

__global__ void fuse_kernel(float* __restrict__ fused, const float* __restrict__ outs,
                            const float* __restrict__ w) {
  int i = blockIdx.x * 256 + threadIdx.x;              // N_*D_ threads
  const size_t ND = (size_t)N_ * D_;
  fused[i] = w[0] * outs[i] + w[1] * outs[ND + i] + w[2] * outs[2 * ND + i] + w[3] * outs[3 * ND + i];
}

// ---------------------------------------------------------------------------
static void launch_gemm(hipStream_t st, const u16* A, int lda, const u16* B, int ldb,
                        float* C, int ldc, const float* bias, const float* resid, int ldr,
                        int M, int N, int K, float alpha, int flags) {
  dim3 grid((N + BN - 1) / BN, (M + BM - 1) / BM);
  wmma_gemm<<<grid, dim3(256), 0, st>>>(A, lda, B, ldb, C, ldc, bias, resid, ldr, M, N, K, alpha, flags);
}

extern "C" void kernel_launch(void* const* d_in, const int* in_sizes, int n_in,
                              void* d_out, int out_size, void* d_ws, size_t ws_size,
                              hipStream_t stream)
{
  (void)in_sizes; (void)n_in; (void)out_size; (void)ws_size;
  const float* tokens = (const float*)d_in[0];
  const float* Wqkv   = (const float*)d_in[1];
  const float* bqkv   = (const float*)d_in[2];
  const float* Wo     = (const float*)d_in[3];
  const float* bo     = (const float*)d_in[4];
  const float* ln1s   = (const float*)d_in[5];
  const float* ln1b   = (const float*)d_in[6];
  const float* ln2s   = (const float*)d_in[7];
  const float* ln2b   = (const float*)d_in[8];
  const float* W1     = (const float*)d_in[9];
  const float* b1     = (const float*)d_in[10];
  const float* W2     = (const float*)d_in[11];
  const float* b2     = (const float*)d_in[12];
  const float* mem    = (const float*)d_in[13];
  const float* Wagg   = (const float*)d_in[14];
  const float* bagg   = (const float*)d_in[15];
  const float* gate   = (const float*)d_in[16];

  char* ws = (char*)d_ws;
  size_t off = 0;
  auto alloc = [&](size_t bytes) -> char* {
    char* p = ws + off;
    off += (bytes + 255) & ~(size_t)255;
    return p;
  };

  // bf16 weight copies
  u16* wqkv_bf = (u16*)alloc((size_t)E_ * 3 * D_ * D_ * 2);
  u16* wo_bf   = (u16*)alloc((size_t)E_ * D_ * D_ * 2);
  u16* w1_bf   = (u16*)alloc((size_t)E_ * DFF_ * D_ * 2);
  u16* w2_bf   = (u16*)alloc((size_t)E_ * D_ * DFF_ * 2);
  u16* wagg_bf = (u16*)alloc((size_t)D_ * L_ * D_ * 2);
  u16* mem_bf  = (u16*)alloc((size_t)L_ * M_ * D_ * 2);
  u16* memT_bf = (u16*)alloc((size_t)L_ * D_ * M_ * 2);
  // per-expert scratch (reused across experts)
  float* t_f    = (float*)alloc((size_t)N_ * D_ * 4);
  u16*   t_bf   = (u16*)  alloc((size_t)N_ * D_ * 2);
  float* cs_f   = (float*)alloc((size_t)CTX_ * D_ * 4);
  u16*   cs_bf  = (u16*)  alloc((size_t)128 * D_ * 2);      // padded to 128 B-rows
  u16*   csT_bf = (u16*)  alloc((size_t)D_ * CTX_ * 2);
  float* clog_f = (float*)alloc((size_t)N_ * CTX_ * 4);
  u16*   p64_bf = (u16*)  alloc((size_t)N_ * CTX_ * 2);
  float* x_f    = (float*)alloc((size_t)N_ * D_ * 4);
  u16*   x_bf   = (u16*)  alloc((size_t)N_ * D_ * 2);
  float* qkv_f  = (float*)alloc((size_t)N_ * 3 * D_ * 4);   // reused later as retr_f
  u16*   qh_bf  = (u16*)  alloc((size_t)N_ * DH_ * 2);
  u16*   kh_bf  = (u16*)  alloc((size_t)N_ * DH_ * 2);
  u16*   vt_bf  = (u16*)  alloc((size_t)128 * N_ * 2);      // padded to 128 B-rows
  float* S_f    = (float*)alloc((size_t)N_ * N_ * 4);       // scores / FFN hidden / mem logits
  u16*   P_bf   = (u16*)  alloc((size_t)N_ * N_ * 2);       // probs / FFN hidden bf16
  float* o_f    = (float*)alloc((size_t)N_ * D_ * 4);
  u16*   o_bf   = (u16*)  alloc((size_t)N_ * D_ * 2);
  float* tmp_f  = (float*)alloc((size_t)N_ * D_ * 4);
  float* x1_f   = (float*)alloc((size_t)N_ * D_ * 4);
  u16*   x1_bf  = (u16*)  alloc((size_t)N_ * D_ * 2);
  float* y_f    = (float*)alloc((size_t)N_ * D_ * 4);
  u16*   y_bf   = (u16*)  alloc((size_t)N_ * D_ * 2);
  u16*   retr_bf= (u16*)  alloc((size_t)N_ * L_ * D_ * 2);
  float* outs   = (float*)alloc((size_t)E_ * N_ * D_ * 4);
  float* pooled = (float*)alloc((size_t)E_ * D_ * 4);
  float* w_ws   = (float*)alloc(64);

  float* retr_f = qkv_f;                  // disjoint lifetimes
  float* fused  = (float*)d_out;
  float* w_out  = fused + (size_t)N_ * D_;

  const float INV_TEMP = 0.044194173824159216f;  // 1/sqrt(512)

  auto cvtN = [&](u16* dst, const float* src, size_t n) {
    cvt_bf16<<<dim3((unsigned)((n + 255) / 256)), dim3(256), 0, stream>>>(dst, src, (int)n);
  };

  // --- per-launch weight conversion ---
  cvtN(wqkv_bf, Wqkv, (size_t)E_ * 3 * D_ * D_);
  cvtN(wo_bf,   Wo,   (size_t)E_ * D_ * D_);
  cvtN(w1_bf,   W1,   (size_t)E_ * DFF_ * D_);
  cvtN(w2_bf,   W2,   (size_t)E_ * D_ * DFF_);
  cvtN(wagg_bf, Wagg, (size_t)D_ * L_ * D_);
  cvtN(mem_bf,  mem,  (size_t)L_ * M_ * D_);
  for (int l = 0; l < L_; ++l)
    transpose_pack<<<dim3((M_ * D_ + 255) / 256), 256, 0, stream>>>(
        memT_bf + (size_t)l * D_ * M_, mem + (size_t)l * M_ * D_, M_, D_, D_);

  for (int e = 0; e < E_; ++e) {
    ln_rows<<<N_, 256, 0, stream>>>(t_f, t_bf, tokens + (size_t)e * N_ * D_, nullptr, nullptr);
    const float* ctx = tokens + (size_t)((e == 0) ? 1 : 0) * N_ * D_;
    ln_rows<<<CTX_, 256, 0, stream>>>(cs_f, cs_bf, ctx, nullptr, nullptr);
    transpose_pack<<<dim3((CTX_ * D_ + 255) / 256), 256, 0, stream>>>(csT_bf, cs_f, CTX_, D_, D_);

    // context attention: clog = clip(t @ cs^T / temp); x = t + softmax(clog) @ cs
    launch_gemm(stream, t_bf, D_, cs_bf, D_, clog_f, CTX_, nullptr, nullptr, 0,
                N_, CTX_, D_, INV_TEMP, F_CLIP);
    softmax_rows<<<N_, 256, 0, stream>>>(p64_bf, clog_f, CTX_);
    launch_gemm(stream, p64_bf, CTX_, csT_bf, CTX_, x_f, D_, nullptr, t_f, D_,
                N_, D_, CTX_, 1.0f, 0);
    cvtN(x_bf, x_f, (size_t)N_ * D_);

    // qkv = x @ Wqkv^T + bqkv
    launch_gemm(stream, x_bf, D_, wqkv_bf + (size_t)e * 3 * D_ * D_, D_, qkv_f, 3 * D_,
                bqkv + (size_t)e * 3 * D_, nullptr, 0, N_, 3 * D_, D_, 1.0f, 0);

    // 8-head attention
    for (int h = 0; h < H_; ++h) {
      pack_cols<<<dim3((N_ * DH_ + 255) / 256), 256, 0, stream>>>(qh_bf, qkv_f + h * DH_, N_, DH_, 3 * D_);
      pack_cols<<<dim3((N_ * DH_ + 255) / 256), 256, 0, stream>>>(kh_bf, qkv_f + D_ + h * DH_, N_, DH_, 3 * D_);
      transpose_pack<<<dim3((N_ * DH_ + 255) / 256), 256, 0, stream>>>(vt_bf, qkv_f + 2 * D_ + h * DH_, N_, DH_, 3 * D_);
      launch_gemm(stream, qh_bf, DH_, kh_bf, DH_, S_f, N_, nullptr, nullptr, 0,
                  N_, N_, DH_, 0.125f, 0);
      softmax_rows<<<N_, 256, 0, stream>>>(P_bf, S_f, N_);
      launch_gemm(stream, P_bf, N_, vt_bf, N_, o_f + h * DH_, D_, nullptr, nullptr, 0,
                  N_, DH_, N_, 1.0f, 0);
    }
    cvtN(o_bf, o_f, (size_t)N_ * D_);

    // x1 = LN1(x + o @ Wo^T + bo)
    launch_gemm(stream, o_bf, D_, wo_bf + (size_t)e * D_ * D_, D_, tmp_f, D_,
                bo + (size_t)e * D_, x_f, D_, N_, D_, D_, 1.0f, 0);
    ln_rows<<<N_, 256, 0, stream>>>(x1_f, x1_bf, tmp_f, ln1s + (size_t)e * D_, ln1b + (size_t)e * D_);

    // FFN: y = LN2(x1 + relu(x1@W1^T+b1)@W2^T + b2)
    launch_gemm(stream, x1_bf, D_, w1_bf + (size_t)e * DFF_ * D_, D_, S_f, DFF_,
                b1 + (size_t)e * DFF_, nullptr, 0, N_, DFF_, D_, 1.0f, F_RELU);
    cvtN(P_bf, S_f, (size_t)N_ * DFF_);
    launch_gemm(stream, P_bf, DFF_, w2_bf + (size_t)e * D_ * DFF_, DFF_, tmp_f, D_,
                b2 + (size_t)e * D_, x1_f, D_, N_, D_, DFF_, 1.0f, 0);
    ln_rows<<<N_, 256, 0, stream>>>(y_f, y_bf, tmp_f, ln2s + (size_t)e * D_, ln2b + (size_t)e * D_);

    // memory retrieval (3 layers)
    for (int l = 0; l < L_; ++l) {
      launch_gemm(stream, y_bf, D_, mem_bf + (size_t)l * M_ * D_, D_, S_f, M_,
                  nullptr, nullptr, 0, N_, M_, D_, INV_TEMP, F_CLIP);
      softmax_rows<<<N_, 256, 0, stream>>>(P_bf, S_f, M_);
      launch_gemm(stream, P_bf, M_, memT_bf + (size_t)l * D_ * M_, M_, retr_f + l * D_, L_ * D_,
                  nullptr, nullptr, 0, N_, D_, M_, 1.0f, 0);
    }
    cvtN(retr_bf, retr_f, (size_t)N_ * L_ * D_);

    // outs[e] = y + retr @ Wagg^T + bagg
    launch_gemm(stream, retr_bf, L_ * D_, wagg_bf, L_ * D_, outs + (size_t)e * N_ * D_, D_,
                bagg, y_f, D_, N_, D_, L_ * D_, 1.0f, 0);
  }

  // gating + fusion
  mean_pool<<<dim3((E_ * D_) / 256), 256, 0, stream>>>(pooled, outs);
  gate_kernel<<<1, 256, 0, stream>>>(w_ws, w_out, pooled, gate);
  fuse_kernel<<<dim3((N_ * D_) / 256), 256, 0, stream>>>(fused, outs, w_ws);
}
